// TorchViTSelfAttention_77197742179020
// MI455X (gfx1250) — matrix operand order repaired and checked by reading, compiled, not run on hardware
//
#include <hip/hip_runtime.h>
#include <hip/hip_bf16.h>
#include <stdint.h>

// ---------------------------------------------------------------------------
// Types for CDNA5 WMMA (gfx1250, wave32):  v_wmma_f32_16x16x32_bf16
// ---------------------------------------------------------------------------
typedef __bf16 bf16;
typedef __attribute__((ext_vector_type(8)))  __bf16 v8bf;
typedef __attribute__((ext_vector_type(16))) __bf16 v16bf;
typedef __attribute__((ext_vector_type(8)))  float  v8f;
typedef __attribute__((ext_vector_type(4)))  int    v4i;

union V16 { v16bf v; v8bf h[2]; };

__device__ __forceinline__ bf16 f2bf(float f) {
  // round-to-nearest-even fp32 -> bf16
  unsigned u = __builtin_bit_cast(unsigned, f);
  unsigned r = u + 0x7FFFu + ((u >> 16) & 1u);
  unsigned short s = (unsigned short)(r >> 16);
  return __builtin_bit_cast(bf16, s);
}

// ---------------------------------------------------------------------------
// Async global->LDS copy (gfx1250 GLOBAL_LOAD_ASYNC_TO_LDS_B128, ASYNCcnt).
// The builtin's pointer params are int4* in AS1 (global) / AS3 (LDS).
// Guarded by __has_builtin; falls back to a synchronous VGPR round-trip.
// ---------------------------------------------------------------------------
#if defined(__AMDGCN__) && __has_builtin(__builtin_amdgcn_global_load_async_to_lds_b128)
#define HAVE_ASYNC_LDS 1
typedef __attribute__((address_space(1))) v4i gas_v4i;
typedef __attribute__((address_space(3))) v4i las_v4i;
#else
#define HAVE_ASYNC_LDS 0
#endif

__device__ __forceinline__ void copy16_g2lds(bf16* lds, const bf16* g) {
#if HAVE_ASYNC_LDS
  __builtin_amdgcn_global_load_async_to_lds_b128(
      (gas_v4i*)g, (las_v4i*)lds, 0, 0);
#else
  *(v8bf*)lds = *(const v8bf*)g;
#endif
}

__device__ __forceinline__ void wait_async_copies() {
#if HAVE_ASYNC_LDS
#if __has_builtin(__builtin_amdgcn_s_wait_asynccnt)
  __builtin_amdgcn_s_wait_asynccnt(0);
#else
  asm volatile("s_wait_asynccnt 0" ::: "memory");
#endif
#endif
}

// Problem constants
#define BATCH   16
#define SEQ     1024
#define CDIM    768
#define HEADS   12
#define DHEAD   64
#define BNROWS  (BATCH * SEQ)          // 16384

// ---------------------------------------------------------------------------
// fp32 -> bf16 conversion (grid-stride, non-temporal reads)
// ---------------------------------------------------------------------------
__global__ void cvt_bf16_kernel(const float* __restrict__ src,
                                bf16* __restrict__ dst, int n) {
  int i = blockIdx.x * blockDim.x + threadIdx.x;
  int stride = gridDim.x * blockDim.x;
  for (; i < n; i += stride) {
    float v = __builtin_nontemporal_load(&src[i]);
    dst[i] = f2bf(v);
  }
}

// ---------------------------------------------------------------------------
// Tiled GEMM: Y(MxN) = X(MxK, bf16 row-major) @ W(KxN, bf16 row-major) + bias
// Block tile 128x256, BK=32, double-buffered LDS. 8 waves: 2 (M) x 4 (N),
// each wave 64x64 = 4x4 WMMA tiles -> 16 v_wmma per k-step per wave.
// A tile copied global->LDS via async DMA (ASYNCcnt); B tile staged through
// VGPRs (transposed to N-major so B fragments are two contiguous
// ds_read_b128s matching the ISA B layout), with its global loads issued
// before the WMMAs and its ds_stores after the fragment ds_loads so the
// in-order DS queue never blocks compute.
// ---------------------------------------------------------------------------
template <bool OUT_F32>
__global__ __launch_bounds__(256)
void gemm_bias_kernel(const bf16* __restrict__ X, const bf16* __restrict__ W,
                      const float* __restrict__ bias, void* __restrict__ Yv,
                      int M, int N, int K) {
  constexpr int BM = 128, BNt = 256, BK = 32;
  constexpr int LDA = BK + 8;  // 40 bf16 = 80B row stride (16B aligned)
  __shared__ __align__(16) bf16 As[2][BM * LDA];    // [row][k]
  __shared__ __align__(16) bf16 Bs[2][BNt * LDA];   // [n][k] (transposed)

  const int tid  = threadIdx.x;
  const int lane = tid & 31;
  const int wave = tid >> 5;
  const int wm   = wave & 1;   // 0..1 -> 64-row slab
  const int wn   = wave >> 1;  // 0..3 -> 64-col slab
  const int l16  = lane & 15;
  const int lh   = lane >> 4;
  const int bm   = blockIdx.x * BM;
  const int bn   = blockIdx.y * BNt;

  // copier thread mappings
  const int ar = tid >> 2, ac = (tid & 3) * 8;    // A: 128x32 in 2 b128/thr
  const int bkr = tid >> 5, bnc = (tid & 31) * 8; // B: 32x256 in 4 v8bf/thr

  v8f acc[4][4];
#pragma unroll
  for (int i = 0; i < 4; ++i)
#pragma unroll
    for (int j = 0; j < 4; ++j) acc[i][j] = (v8f)0.0f;

  // ---- prologue: stage tile k0=0 into buffer 0 ----
  {
    const bf16* src = X + (size_t)(bm + ar) * K + ac;
    copy16_g2lds(&As[0][ar * LDA + ac], src);
    copy16_g2lds(&As[0][(ar + 64) * LDA + ac], src + (size_t)64 * K);
#pragma unroll
    for (int i = 0; i < 4; ++i) {
      int kk = bkr + i * 8;
      v8bf t = *(const v8bf*)(W + (size_t)kk * N + bn + bnc);
#pragma unroll
      for (int e = 0; e < 8; ++e) Bs[0][(bnc + e) * LDA + kk] = t[e];
    }
  }
  wait_async_copies();
  __syncthreads();

  int buf = 0;
  for (int k0 = 0; k0 < K; k0 += BK) {
    const bool has_next = (k0 + BK) < K;
    v8bf bt[4];
    if (has_next) {
      // B global loads for tile k0+BK (latency hidden by the WMMAs below)
#pragma unroll
      for (int i = 0; i < 4; ++i)
        bt[i] = *(const v8bf*)(W + (size_t)(k0 + BK + bkr + i * 8) * N + bn + bnc);
      // A async DMA prefetch into the other buffer
      const bf16* src = X + (size_t)(bm + ar) * K + (k0 + BK) + ac;
      copy16_g2lds(&As[buf ^ 1][ar * LDA + ac], src);
      copy16_g2lds(&As[buf ^ 1][(ar + 64) * LDA + ac], src + (size_t)64 * K);
    }

    // ---- fragments + 16 WMMAs from the current buffer ----
    const bf16* Ac = As[buf];
    const bf16* Bc = Bs[buf];
    V16 a[4], b[4];
#pragma unroll
    for (int mt = 0; mt < 4; ++mt) {
      const bf16* base = Ac + (wm * 64 + mt * 16 + l16) * LDA;  // lane = M row
      a[mt].h[0] = *(const v8bf*)(base + lh * 8);       // K 0-7 / 8-15
      a[mt].h[1] = *(const v8bf*)(base + 16 + lh * 8);  // K 16-23 / 24-31
    }
#pragma unroll
    for (int nt = 0; nt < 4; ++nt) {
      const bf16* base = Bc + (wn * 64 + nt * 16 + l16) * LDA;  // lane = N col
      b[nt].h[0] = *(const v8bf*)(base + lh * 8);       // K 0-7 / 8-15
      b[nt].h[1] = *(const v8bf*)(base + 16 + lh * 8);  // K 16-23 / 24-31
    }
#pragma unroll
    for (int mt = 0; mt < 4; ++mt)
#pragma unroll
      for (int nt = 0; nt < 4; ++nt)
        acc[mt][nt] = __builtin_amdgcn_wmma_f32_16x16x32_bf16(
            false, a[mt].v, false, b[nt].v, (short)0, acc[mt][nt], false, false);

    // ---- B transpose store for the next tile (after fragment ds_loads) ----
    if (has_next) {
#pragma unroll
      for (int i = 0; i < 4; ++i)
#pragma unroll
        for (int e = 0; e < 8; ++e)
          Bs[buf ^ 1][(bnc + e) * LDA + bkr + i * 8] = bt[i][e];
    }
    wait_async_copies();
    __syncthreads();
    buf ^= 1;
  }

  // ---- epilogue: C/D layout (lane = N, rows = lane-half*8 + r) ----
#pragma unroll
  for (int mt = 0; mt < 4; ++mt) {
#pragma unroll
    for (int nt = 0; nt < 4; ++nt) {
      int col = bn + wn * 64 + nt * 16 + l16;
      float bv = bias[col];
#pragma unroll
      for (int r = 0; r < 8; ++r) {
        int row = bm + wm * 64 + mt * 16 + lh * 8 + r;
        float val = acc[mt][nt][r] + bv;
        if (OUT_F32)
          __builtin_nontemporal_store(val, &((float*)Yv)[(size_t)row * N + col]);
        else
          ((bf16*)Yv)[(size_t)row * N + col] = f2bf(val);
      }
    }
  }
}

// ---------------------------------------------------------------------------
// Flash-style attention over flat-reshaped heads, double-buffered K/V tiles.
// q2/k2/v2 are the *flat* reshape of (B,N,C) -> (B*H, N, Dh): group g reads
// q/k/v at flat offset g*N*Dh. The output side does the real transpose:
// ctx[b, n, h*64+d] = ctx2[g=b*H+h, n, d], fused into the bf16 store to X2.
// Block: 256 thr = 8 waves; wave handles 16 query rows; 64-key blocks.
// K tile is copied by async DMA; V is staged through VGPRs (transposed).
// ---------------------------------------------------------------------------
__global__ __launch_bounds__(256)
void attention_kernel(const bf16* __restrict__ Qb, const bf16* __restrict__ Kb,
                      const bf16* __restrict__ Vb, bf16* __restrict__ X2) {
  constexpr int KB = 64;          // keys per block iteration
  constexpr int LD = DHEAD + 8;   // 72 bf16 row stride (144B, 16B aligned)
  __shared__ __align__(16) bf16 Ks[2][KB * LD];     // [key][d]
  __shared__ __align__(16) bf16 Vt[2][DHEAD * LD];  // [d][key] (transposed)
  __shared__ __align__(16) bf16 Ps[8 * 16 * LD];    // per-wave P tile [16][64]

  const int g    = blockIdx.x >> 3;        // 0..191 head-group
  const int q0   = (blockIdx.x & 7) * 128; // query block base
  const int bb   = g / HEADS;
  const int h    = g % HEADS;
  const int tid  = threadIdx.x;
  const int lane = tid & 31;
  const int wave = tid >> 5;
  const int l16  = lane & 15;
  const int lh   = lane >> 4;
  const float scale = 0.125f;  // 1/sqrt(64)

  const bf16* Qg = Qb + (size_t)g * SEQ * DHEAD;
  const bf16* Kg = Kb + (size_t)g * SEQ * DHEAD;
  const bf16* Vg = Vb + (size_t)g * SEQ * DHEAD;

  // copier mapping: 64 rows x 64 d, two v8bf per thread
  const int cr = tid >> 3, cc = (tid & 7) * 8;

  // Q fragments for this wave's 16 rows (A layout: lane = M, halves K+0/+8)
  const int qrow = q0 + wave * 16 + l16;
  V16 qa[2];
#pragma unroll
  for (int s = 0; s < 2; ++s) {
    qa[s].h[0] = *(const v8bf*)&Qg[(size_t)qrow * DHEAD + s * 32 + lh * 8];
    qa[s].h[1] = *(const v8bf*)&Qg[(size_t)qrow * DHEAD + s * 32 + 16 + lh * 8];
  }

  float m_i[8], l_i[8];
  v8f acc_o[4];
#pragma unroll
  for (int r = 0; r < 8; ++r) { m_i[r] = -3.0e38f; l_i[r] = 0.0f; }
#pragma unroll
  for (int dt = 0; dt < 4; ++dt) acc_o[dt] = (v8f)0.0f;

  bf16* Pw = Ps + wave * 16 * LD;

  // ---- prologue: stage key block 0 into buffer 0 ----
#pragma unroll
  for (int i = 0; i < 2; ++i) {
    int kk = cr + i * 32;
    copy16_g2lds(&Ks[0][kk * LD + cc], &Kg[(size_t)kk * DHEAD + cc]);
    v8bf vv = *(const v8bf*)&Vg[(size_t)kk * DHEAD + cc];
#pragma unroll
    for (int e = 0; e < 8; ++e) Vt[0][(cc + e) * LD + kk] = vv[e];
  }
  wait_async_copies();
  __syncthreads();

  int buf = 0;
  for (int j0 = 0; j0 < SEQ; j0 += KB) {
    const bool has_next = (j0 + KB) < SEQ;
    v8bf vnx[2];
    if (has_next) {
#pragma unroll
      for (int i = 0; i < 2; ++i) {
        int kk = cr + i * 32;
        copy16_g2lds(&Ks[buf ^ 1][kk * LD + cc],
                     &Kg[(size_t)(j0 + KB + kk) * DHEAD + cc]);
        vnx[i] = *(const v8bf*)&Vg[(size_t)(j0 + KB + kk) * DHEAD + cc];
      }
    }
    const bf16* Kc = Ks[buf];
    const bf16* Vc = Vt[buf];

    // --- S = (Q @ K^T) * scale : 4 key-tiles x 2 k-steps = 8 WMMAs ---
    v8f sa[4];
#pragma unroll
    for (int nt = 0; nt < 4; ++nt) sa[nt] = (v8f)0.0f;
#pragma unroll
    for (int nt = 0; nt < 4; ++nt) {
      const bf16* base = Kc + (nt * 16 + l16) * LD;  // B layout: lane = key
#pragma unroll
      for (int s = 0; s < 2; ++s) {
        V16 bf;
        bf.h[0] = *(const v8bf*)(base + s * 32 + lh * 8);
        bf.h[1] = *(const v8bf*)(base + s * 32 + 16 + lh * 8);
        sa[nt] = __builtin_amdgcn_wmma_f32_16x16x32_bf16(
            false, qa[s].v, false, bf.v, (short)0, sa[nt], false, false);
      }
    }

    // --- online softmax (rows live at lane-half*8+r; reduce over 16 lanes) ---
#pragma unroll
    for (int r = 0; r < 8; ++r) {
      float mx = -3.0e38f;
#pragma unroll
      for (int nt = 0; nt < 4; ++nt) {
        sa[nt][r] *= scale;
        mx = fmaxf(mx, sa[nt][r]);
      }
#pragma unroll
      for (int off = 1; off < 16; off <<= 1)
        mx = fmaxf(mx, __shfl_xor(mx, off, 32));
      float mn = fmaxf(m_i[r], mx);
      float corr = __expf(m_i[r] - mn);
      l_i[r] *= corr;
#pragma unroll
      for (int dt = 0; dt < 4; ++dt) acc_o[dt][r] *= corr;
      m_i[r] = mn;
      float rs = 0.0f;
#pragma unroll
      for (int nt = 0; nt < 4; ++nt) {
        float p = __expf(sa[nt][r] - mn);
        rs += p;
        Pw[(lh * 8 + r) * LD + nt * 16 + l16] = f2bf(p);  // C-layout scatter
      }
#pragma unroll
      for (int off = 1; off < 16; off <<= 1) rs += __shfl_xor(rs, off, 32);
      l_i[r] += rs;
    }

    // --- ctx += P @ V : P read back in A layout, V^T in B layout. 8 WMMAs ---
    V16 pf[2];
#pragma unroll
    for (int s = 0; s < 2; ++s) {
      pf[s].h[0] = *(const v8bf*)&Pw[l16 * LD + s * 32 + lh * 8];
      pf[s].h[1] = *(const v8bf*)&Pw[l16 * LD + s * 32 + 16 + lh * 8];
    }
#pragma unroll
    for (int dt = 0; dt < 4; ++dt) {
      const bf16* base = Vc + (dt * 16 + l16) * LD;  // B layout: lane = d
#pragma unroll
      for (int s = 0; s < 2; ++s) {
        V16 vf;
        vf.h[0] = *(const v8bf*)(base + s * 32 + lh * 8);
        vf.h[1] = *(const v8bf*)(base + s * 32 + 16 + lh * 8);
        acc_o[dt] = __builtin_amdgcn_wmma_f32_16x16x32_bf16(
            false, pf[s].v, false, vf.v, (short)0, acc_o[dt], false, false);
      }
    }

    // --- V transpose store for next tile (after PV fragment ds_loads) ---
    if (has_next) {
#pragma unroll
      for (int i = 0; i < 2; ++i) {
        int kk = cr + i * 32;
#pragma unroll
        for (int e = 0; e < 8; ++e) Vt[buf ^ 1][(cc + e) * LD + kk] = vnx[i][e];
      }
    }
    wait_async_copies();
    __syncthreads();
    buf ^= 1;
  }

  // --- normalize + transposed store into (B,N,C) bf16 for final projection ---
#pragma unroll
  for (int dt = 0; dt < 4; ++dt) {
    int c = h * DHEAD + dt * 16 + l16;
#pragma unroll
    for (int r = 0; r < 8; ++r) {
      int row = q0 + wave * 16 + lh * 8 + r;
      float val = acc_o[dt][r] / l_i[r];
      X2[(size_t)(bb * SEQ + row) * CDIM + c] = f2bf(val);
    }
  }
}

// ---------------------------------------------------------------------------
// Launcher
// ---------------------------------------------------------------------------
extern "C" void kernel_launch(void* const* d_in, const int* in_sizes, int n_in,
                              void* d_out, int out_size, void* d_ws, size_t ws_size,
                              hipStream_t stream) {
  const float* x  = (const float*)d_in[0];
  const float* Wq = (const float*)d_in[1];
  const float* bq = (const float*)d_in[2];
  const float* Wk = (const float*)d_in[3];
  const float* bk = (const float*)d_in[4];
  const float* Wv = (const float*)d_in[5];
  const float* bv = (const float*)d_in[6];
  const float* Wo = (const float*)d_in[7];
  const float* bo = (const float*)d_in[8];

  const size_t XN = (size_t)BNROWS * CDIM;   // 12,582,912
  const size_t WN = (size_t)CDIM * CDIM;     // 589,824

  bf16* xb  = (bf16*)d_ws;
  bf16* qb  = xb  + XN;
  bf16* kb  = qb  + XN;
  bf16* vb  = kb  + XN;
  bf16* x2  = vb  + XN;
  bf16* wqb = x2  + XN;
  bf16* wkb = wqb + WN;
  bf16* wvb = wkb + WN;
  bf16* wob = wvb + WN;

  // fp32 -> bf16 staging
  cvt_bf16_kernel<<<2048, 256, 0, stream>>>(x,  xb,  (int)XN);
  cvt_bf16_kernel<<<576,  256, 0, stream>>>(Wq, wqb, (int)WN);
  cvt_bf16_kernel<<<576,  256, 0, stream>>>(Wk, wkb, (int)WN);
  cvt_bf16_kernel<<<576,  256, 0, stream>>>(Wv, wvb, (int)WN);
  cvt_bf16_kernel<<<576,  256, 0, stream>>>(Wo, wob, (int)WN);

  // Q/K/V projections (bf16 out), block tile 128x256
  dim3 gproj(BNROWS / 128, CDIM / 256);  // (128, 3)
  gemm_bias_kernel<false><<<gproj, 256, 0, stream>>>(xb, wqb, bq, (void*)qb,
                                                     BNROWS, CDIM, CDIM);
  gemm_bias_kernel<false><<<gproj, 256, 0, stream>>>(xb, wkb, bk, (void*)kb,
                                                     BNROWS, CDIM, CDIM);
  gemm_bias_kernel<false><<<gproj, 256, 0, stream>>>(xb, wvb, bv, (void*)vb,
                                                     BNROWS, CDIM, CDIM);

  // attention: 192 groups x 8 query blocks
  attention_kernel<<<BATCH * HEADS * (SEQ / 128), 256, 0, stream>>>(qb, kb, vb, x2);

  // output projection (fp32 out + bias)
  gemm_bias_kernel<true><<<gproj, 256, 0, stream>>>(x2, wob, bo, d_out,
                                                    BNROWS, CDIM, CDIM);
}
